// GCN_37769942401081
// MI455X (gfx1250) — matrix operand order repaired and checked by reading, not compiled, not run on hardware
//
#include <hip/hip_runtime.h>
#include <hip/hip_bf16.h>

// ---------------------------------------------------------------------------
// GCN (3x GCNConv + BN + ReLU, 2x TopKPooling) for MI455X / gfx1250.
//  * GEMMs: split-bf16 (bf16x3) via v_wmma_f32_16x16x32_bf16, fp32 accum.
//    W pre-packed per conv into WMMA B-fragment layout (hi/lo bf16) so the
//    inner loop does one 32B vector load per fragment, no conversion.
//  * Aggregation: wave-per-edge float4 gather + global f32 atomics (L2-resident).
//  * TopK: bucketed exact rank (matches jax.lax.top_k tie rule), ~O(N).
// ---------------------------------------------------------------------------

#define NN      50000
#define NE      800000
#define CC      128
#define KK1     37500
#define KK1P    37504      // padded to a multiple of 16 for WMMA tiles
#define KK2     28125
#define BN_EPS  1e-5f
#define NBUCKET 65536

typedef __attribute__((ext_vector_type(16))) __bf16 v16bf;
typedef __attribute__((ext_vector_type(8)))  float  v8f;

static __device__ __forceinline__ __bf16 bf16_from_bits(unsigned short u) {
    union { unsigned short s; __bf16 h; } x; x.s = u; return x.h;
}
static __device__ __forceinline__ float bf16_to_float(__bf16 h) {
    union { unsigned short s; __bf16 h; } x; x.h = h;
    return __uint_as_float(((unsigned)x.s) << 16);
}
static __device__ __forceinline__ __bf16 f32_to_bf16_rtz(float f) {
    return bf16_from_bits((unsigned short)(__float_as_uint(f) >> 16));
}
// monotonic (ascending) u32 key for float compare; -0 canonicalized by caller
static __device__ __forceinline__ unsigned f32_sort_key(float f) {
    unsigned u = __float_as_uint(f);
    return (u & 0x80000000u) ? ~u : (u | 0x80000000u);
}

// ---------------------------------------------------------------------------
// Pack W [128,128] f32 row-major into split-bf16 WMMA B-fragments.
// Fragment f = nt*4+kc covers cols nt*16..+15, K rows kc*32..+31.
// Per ISA 7.12.2: lane col = lane&15; lanes 0-15 hold K k0+{0..7,16..23},
// lanes 16-31 hold K k0+{8..15,24..31} (16 bf16 per lane).
// ---------------------------------------------------------------------------
__global__ void GCN_prep_w(const float* __restrict__ Wm,
                           __bf16* __restrict__ WH, __bf16* __restrict__ WL)
{
    int t = blockIdx.x * blockDim.x + threadIdx.x;       // 0..1023
    if (t >= 32 * 32) return;
    int f = t >> 5, lane = t & 31;
    int nt = f >> 2, kc = f & 3;
    int n0 = nt * 16, k0 = kc * 32;
    int bcol = lane & 15;
    int bks  = (lane >> 4) * 8;
    long base = (long)t * 16;
#pragma unroll
    for (int e = 0; e < 16; ++e) {
        float w = Wm[(long)(k0 + bks + e + ((e >> 3) << 3)) * CC + n0 + bcol];
        __bf16 h = f32_to_bf16_rtz(w);
        WH[base + e] = h;
        WL[base + e] = f32_to_bf16_rtz(w - bf16_to_float(h));
    }
}

// ---------------------------------------------------------------------------
// GEMM: H[m,n] = sum_k X[m,k] * W[k,n];  M = ntiles*16 rows, K = N = 128.
// Split-bf16: X ~ Xh + Xl, W ~ Wh + Wl;  X*W ~ Xh*Wh + Xl*Wh + Xh*Wl.
// One wave per 16-row strip; A fragments (hi/lo x 4 k-chunks) in VGPRs,
// B fragments pre-packed (one 32B load each).
// ---------------------------------------------------------------------------
__global__ void GCN_gemm_bf16x3(const float* __restrict__ X,
                                const __bf16* __restrict__ WH,
                                const __bf16* __restrict__ WL,
                                float* __restrict__ Hout, int ntiles)
{
    int tile = blockIdx.x * (blockDim.x >> 5) + (threadIdx.x >> 5);
    if (tile >= ntiles) return;                 // whole wave exits together
    int lane = threadIdx.x & 31;
    int m0   = tile * 16;

    // ---- A fragments: 16x32 bf16. lane row = m0 + (lane&15);
    //      lanes 0-15 hold K {0..7,16..23}+k0, lanes 16-31 hold K {8..15,24..31}+k0
    int arow = m0 + (lane & 15);
    int akb  = (lane >> 4) * 8;
    const float4* xp = (const float4*)(X + (long)arow * CC);
    v16bf ah[4], al[4];
    for (int kc = 0; kc < 4; ++kc) {
        int k0 = kc * 32;
        float fa[16];
        float4 q0 = xp[(k0 + akb)      >> 2];
        float4 q1 = xp[(k0 + akb + 4)  >> 2];
        float4 q2 = xp[(k0 + akb + 16) >> 2];
        float4 q3 = xp[(k0 + akb + 20) >> 2];
        fa[0] = q0.x;  fa[1] = q0.y;  fa[2]  = q0.z;  fa[3]  = q0.w;
        fa[4] = q1.x;  fa[5] = q1.y;  fa[6]  = q1.z;  fa[7]  = q1.w;
        fa[8] = q2.x;  fa[9] = q2.y;  fa[10] = q2.z;  fa[11] = q2.w;
        fa[12] = q3.x; fa[13] = q3.y; fa[14] = q3.z;  fa[15] = q3.w;
#pragma unroll
        for (int e = 0; e < 16; ++e) {
            __bf16 h = f32_to_bf16_rtz(fa[e]);
            ah[kc][e] = h;
            al[kc][e] = f32_to_bf16_rtz(fa[e] - bf16_to_float(h));
        }
    }

    int crow0 = m0 + (lane >> 4) * 8;           // C/D: M = vgpr + 8*(lane>=16)
    int ccol  = lane & 15;

    for (int nt = 0; nt < 8; ++nt) {
        int n0 = nt * 16;
        v8f acc = {0.f, 0.f, 0.f, 0.f, 0.f, 0.f, 0.f, 0.f};
        for (int kc = 0; kc < 4; ++kc) {
            long fo = (long)(((nt * 4 + kc) * 32 + lane)) << 4;  // *16 bf16
            v16bf bh = *(const v16bf*)(WH + fo);
            v16bf bl = *(const v16bf*)(WL + fo);
            acc = __builtin_amdgcn_wmma_f32_16x16x32_bf16(false, ah[kc], false, bh,
                                                          (short)0, acc, false, false);
            acc = __builtin_amdgcn_wmma_f32_16x16x32_bf16(false, al[kc], false, bh,
                                                          (short)0, acc, false, false);
            acc = __builtin_amdgcn_wmma_f32_16x16x32_bf16(false, ah[kc], false, bl,
                                                          (short)0, acc, false, false);
        }
#pragma unroll
        for (int r = 0; r < 8; ++r)
            Hout[(long)(crow0 + r) * CC + n0 + ccol] = acc[r];
    }
}

// ---------------------------------------------------------------------------
// Utility / elementwise kernels
// ---------------------------------------------------------------------------
__global__ void GCN_fill_f32(float* __restrict__ p, float v, int n)
{
    int i = blockIdx.x * blockDim.x + threadIdx.x;
    if (i < n) p[i] = v;
}
__global__ void GCN_fill_i32(int* __restrict__ p, int v, int n)
{
    int i = blockIdx.x * blockDim.x + threadIdx.x;
    if (i < n) p[i] = v;
}

// degree for conv1: every edge has weight 1 (deg pre-filled with 1.0 self loop)
__global__ void GCN_deg1(const int* __restrict__ col, float* __restrict__ deg, int E)
{
    int e = blockIdx.x * blockDim.x + threadIdx.x;
    if (e < E) atomicAdd(&deg[col[e]], 1.0f);
}

// degree for pooled graph: edge survives iff both endpoints kept (rank < K)
__global__ void GCN_deg2(const int* __restrict__ row, const int* __restrict__ col,
                         const int* __restrict__ rank, int K,
                         float* __restrict__ deg, int E)
{
    int e = blockIdx.x * blockDim.x + threadIdx.x;
    if (e >= E) return;
    int rr = rank[row[e]], rc = rank[col[e]];
    if (rr < K && rc < K) atomicAdd(&deg[rc], 1.0f);
}

__global__ void GCN_dinv(float* __restrict__ deg, int n)
{
    int i = blockIdx.x * blockDim.x + threadIdx.x;
    if (i < n) { float d = deg[i]; deg[i] = (d > 0.f) ? rsqrtf(d) : 0.f; }
}

// self-loop term: out[i,:] = H[i,:] * dinv[i]^2
__global__ void GCN_agg_self(const float* __restrict__ H, const float* __restrict__ dinv,
                             float* __restrict__ out, int n)
{
    long idx = (long)blockIdx.x * blockDim.x + threadIdx.x;
    if (idx >= (long)n * CC) return;
    int i = (int)(idx >> 7);
    float di = dinv[i];
    out[idx] = H[idx] * di * di;
}

// conv1 edges: wave per edge, lane moves float4 (4 channels)
__global__ void GCN_agg_edges1(const int* __restrict__ row, const int* __restrict__ col,
                               const float* __restrict__ H, const float* __restrict__ dinv,
                               float* __restrict__ out, int E)
{
    long t = (long)blockIdx.x * blockDim.x + threadIdx.x;
    int  e = (int)(t >> 5);
    if (e >= E) return;
    int lane = (int)(t & 31);
    int r = row[e], c = col[e];
    float coef = dinv[r] * dinv[c];
    const float4* hp = (const float4*)(H + (long)r * CC);
    float4 v = hp[lane];
    float* op = out + (long)c * CC + lane * 4;
    atomicAdd(op + 0, v.x * coef);
    atomicAdd(op + 1, v.y * coef);
    atomicAdd(op + 2, v.z * coef);
    atomicAdd(op + 3, v.w * coef);
}

// pooled-graph edges: remap via rank on the fly; skip dropped edges
__global__ void GCN_agg_edges2(const int* __restrict__ row, const int* __restrict__ col,
                               const int* __restrict__ rank, int K,
                               const float* __restrict__ H, const float* __restrict__ dinv,
                               float* __restrict__ out, int E)
{
    long t = (long)blockIdx.x * blockDim.x + threadIdx.x;
    int  e = (int)(t >> 5);
    if (e >= E) return;
    int lane = (int)(t & 31);
    int rr = rank[row[e]], rc = rank[col[e]];
    if (rr >= K || rc >= K) return;             // uniform across the wave
    float coef = dinv[rr] * dinv[rc];
    const float4* hp = (const float4*)(H + (long)rr * CC);
    float4 v = hp[lane];
    float* op = out + (long)rc * CC + lane * 4;
    atomicAdd(op + 0, v.x * coef);
    atomicAdd(op + 1, v.y * coef);
    atomicAdd(op + 2, v.z * coef);
    atomicAdd(op + 3, v.w * coef);
}

// y = relu((agg + b - m) * g * rsqrt(v+eps) + be); zero the padded rows
__global__ void GCN_bias_bn_relu(const float* agg, float* dst,
                                 const float* __restrict__ b, const float* __restrict__ g,
                                 const float* __restrict__ be, const float* __restrict__ m,
                                 const float* __restrict__ v, int n, int npad)
{
    long idx = (long)blockIdx.x * blockDim.x + threadIdx.x;
    if (idx >= (long)npad * CC) return;
    int i = (int)(idx >> 7), c = (int)(idx & 127);
    if (i < n) {
        float scale = g[c] * rsqrtf(v[c] + BN_EPS);
        float y = (agg[idx] + b[c] - m[c]) * scale + be[c];
        dst[idx] = fmaxf(y, 0.f);
    } else {
        dst[idx] = 0.f;
    }
}

// inv_norm = rsqrt(sum p^2), one wave
__global__ void GCN_pnorm(const float* __restrict__ p, float* __restrict__ out)
{
    int lane = threadIdx.x;
    float s = 0.f;
    for (int c = lane; c < CC; c += 32) s += p[c] * p[c];
#pragma unroll
    for (int off = 16; off; off >>= 1) s += __shfl_down(s, off, 32);
    if (lane == 0) out[0] = rsqrtf(s);
}

// score[i] = tanh(dot(A[i,:], p) * inv_norm); one wave per node. Also emits
// the monotonic sort key (with -0 canonicalized to +0).
__global__ void GCN_scores(const float* __restrict__ A, const float* __restrict__ p,
                           const float* __restrict__ invnorm,
                           float* __restrict__ S, unsigned* __restrict__ K, int n)
{
    long t = (long)blockIdx.x * blockDim.x + threadIdx.x;
    int  i = (int)(t >> 5);
    if (i >= n) return;
    int lane = (int)(t & 31);
    float s = 0.f;
    for (int c = lane; c < CC; c += 32) s += A[(long)i * CC + c] * p[c];
#pragma unroll
    for (int off = 16; off; off >>= 1) s += __shfl_down(s, off, 32);
    if (lane == 0) {
        float sc = tanhf(s * invnorm[0]);
        S[i] = sc;
        K[i] = f32_sort_key(sc == 0.f ? 0.f : sc);
    }
}

// ---------------------------------------------------------------------------
// Bucketed exact descending rank (jax.lax.top_k tie rule: lower index wins).
// ---------------------------------------------------------------------------
__global__ void GCN_hist(const unsigned* __restrict__ K, int* __restrict__ hist, int n)
{
    int i = blockIdx.x * blockDim.x + threadIdx.x;
    if (i < n) atomicAdd(&hist[K[i] >> 16], 1);
}

// gt[b] = #elements in buckets strictly greater than b. One workgroup.
__global__ void GCN_bucket_sufscan(const int* __restrict__ hist, int* __restrict__ gt)
{
    __shared__ int tsum[1024];
    int t = threadIdx.x;                 // 1024 threads, 64 bins each
    int base = t * 64;
    int total = 0;
    for (int k = 0; k < 64; ++k) total += hist[base + k];
    tsum[t] = total;
    __syncthreads();
    if (t == 0) {                        // serial exclusive suffix over 1024 groups
        int run = 0;
        for (int g = 1023; g >= 0; --g) { int h = tsum[g]; tsum[g] = run; run += h; }
    }
    __syncthreads();
    int run = tsum[t];                   // elements in groups after this one
    for (int k = 63; k >= 0; --k) { gt[base + k] = run; run += hist[base + k]; }
}

// counting-sort node ids into their bucket's slot range (order within a
// bucket is irrelevant: rank comparison uses (key, index))
__global__ void GCN_bucket_scatter(const unsigned* __restrict__ K,
                                   const int* __restrict__ gt, const int* __restrict__ hist,
                                   int* __restrict__ cursor, int* __restrict__ blist, int n)
{
    int i = blockIdx.x * blockDim.x + threadIdx.x;
    if (i >= n) return;
    int b = (int)(K[i] >> 16);
    int start = n - gt[b] - hist[b];
    blist[start + atomicAdd(&cursor[b], 1)] = i;
}

__global__ void GCN_rank_fast(const unsigned* __restrict__ K,
                              const int* __restrict__ gt, const int* __restrict__ hist,
                              const int* __restrict__ blist, int* __restrict__ rank, int n)
{
    int i = blockIdx.x * blockDim.x + threadIdx.x;
    if (i >= n) return;
    unsigned ki = K[i];
    int b = (int)(ki >> 16);
    int cnt = gt[b];
    int start = n - cnt - hist[b];
    int len = hist[b];
    for (int t = 0; t < len; ++t) {
        int j = blist[start + t];
        unsigned kj = K[j];
        cnt += (kj > ki) || (kj == ki && j < i);
    }
    rank[i] = cnt;
}

// dst[rank[i],:] = A[i,:] * S[i]  for rank[i] < K  (rows land in sorted order)
__global__ void GCN_pool_scatter(const float* __restrict__ A, const float* __restrict__ S,
                                 const int* __restrict__ rank, int K,
                                 float* __restrict__ dst, int n)
{
    long idx = (long)blockIdx.x * blockDim.x + threadIdx.x;
    if (idx >= (long)n * CC) return;
    int i = (int)(idx >> 7), c = (int)(idx & 127);
    int r = rank[i];
    if (r < K) dst[(long)r * CC + c] = A[idx] * S[i];
}

// ---------------------------------------------------------------------------
// Host-side orchestration
// ---------------------------------------------------------------------------
static inline int cdiv(long a, long b) { return (int)((a + b - 1) / b); }

extern "C" void kernel_launch(void* const* d_in, const int* in_sizes, int n_in,
                              void* d_out, int out_size, void* d_ws, size_t ws_size,
                              hipStream_t stream)
{
    (void)in_sizes; (void)n_in; (void)out_size; (void)ws_size;

    const float* x   = (const float*)d_in[0];
    const int*   ei  = (const int*)d_in[1];
    const int*   row = ei;
    const int*   col = ei + NE;
    const float* W1 = (const float*)d_in[2],  *b1 = (const float*)d_in[3];
    const float* g1 = (const float*)d_in[4],  *be1 = (const float*)d_in[5];
    const float* m1 = (const float*)d_in[6],  *v1 = (const float*)d_in[7];
    const float* W2 = (const float*)d_in[8],  *b2 = (const float*)d_in[9];
    const float* g2 = (const float*)d_in[10], *be2 = (const float*)d_in[11];
    const float* m2 = (const float*)d_in[12], *v2 = (const float*)d_in[13];
    const float* W3 = (const float*)d_in[14], *b3 = (const float*)d_in[15];
    const float* g3 = (const float*)d_in[16], *be3 = (const float*)d_in[17];
    const float* m3 = (const float*)d_in[18], *v3 = (const float*)d_in[19];
    const float* p1 = (const float*)d_in[20], *p2 = (const float*)d_in[21];
    float* out = (float*)d_out;

    // workspace carve (float units; all block starts 16B-aligned)
    float*    H    = (float*)d_ws;               // 6,400,000
    float*    A    = H   + (long)NN * CC;        // 6,400,000
    float*    XN   = A   + (long)NN * CC;        // 4,800,512 (pool1 feats, later A3)
    float*    DEG  = XN  + (long)KK1P * CC;      // 50,000
    float*    S    = DEG + NN;                   // 50,000
    float*    PN   = S   + NN;                   // 2 (+pad 2)
    unsigned* KEY  = (unsigned*)(PN + 4);        // 50,000
    int*      R    = (int*)(KEY + NN);           // 50,000
    int*      HIST = R + NN;                     // 65,536
    int*      GT   = HIST + NBUCKET;             // 65,536
    int*      CUR  = GT + NBUCKET;               // 65,536
    int*      BL   = CUR + NBUCKET;              // 50,000
    __bf16*   WH   = (__bf16*)(BL + NN);         // 16,384 bf16 (32 KB)
    __bf16*   WL   = WH + 32 * 32 * 16;          // 16,384 bf16

    const int B = 256;
    const long NC  = (long)NN * CC;
    const long K1C = (long)KK1P * CC;

    // p-vector inverse norms
    GCN_pnorm<<<1, 32, 0, stream>>>(p1, PN + 0);
    GCN_pnorm<<<1, 32, 0, stream>>>(p2, PN + 1);

    // ---------------- conv1 ----------------
    GCN_prep_w<<<4, 256, 0, stream>>>(W1, WH, WL);
    GCN_gemm_bf16x3<<<cdiv(NN / 16, 8), B, 0, stream>>>(x, WH, WL, H, NN / 16);
    GCN_fill_f32<<<cdiv(NN, B), B, 0, stream>>>(DEG, 1.0f, NN);     // self loops
    GCN_deg1<<<cdiv(NE, B), B, 0, stream>>>(col, DEG, NE);
    GCN_dinv<<<cdiv(NN, B), B, 0, stream>>>(DEG, NN);
    GCN_agg_self<<<cdiv(NC, B), B, 0, stream>>>(H, DEG, A, NN);
    GCN_agg_edges1<<<cdiv((long)NE * 32, B), B, 0, stream>>>(row, col, H, DEG, A, NE);
    GCN_bias_bn_relu<<<cdiv(NC, B), B, 0, stream>>>(A, A, b1, g1, be1, m1, v1, NN, NN);

    // ---------------- pool1 (top-K1, sorted) ----------------
    GCN_scores<<<cdiv((long)NN * 32, B), B, 0, stream>>>(A, p1, PN + 0, S, KEY, NN);
    GCN_fill_i32<<<cdiv(NBUCKET, B), B, 0, stream>>>(HIST, 0, NBUCKET);
    GCN_fill_i32<<<cdiv(NBUCKET, B), B, 0, stream>>>(CUR, 0, NBUCKET);
    GCN_hist<<<cdiv(NN, B), B, 0, stream>>>(KEY, HIST, NN);
    GCN_bucket_sufscan<<<1, 1024, 0, stream>>>(HIST, GT);
    GCN_bucket_scatter<<<cdiv(NN, B), B, 0, stream>>>(KEY, GT, HIST, CUR, BL, NN);
    GCN_rank_fast<<<cdiv(NN, B), B, 0, stream>>>(KEY, GT, HIST, BL, R, NN);
    GCN_pool_scatter<<<cdiv(NC, B), B, 0, stream>>>(A, S, R, KK1, XN, NN);
    GCN_fill_f32<<<cdiv((long)(KK1P - KK1) * CC, B), B, 0, stream>>>(XN + (long)KK1 * CC,
                                                                    0.f, (KK1P - KK1) * CC);

    // ---------------- conv2 (pooled graph) ----------------
    GCN_prep_w<<<4, 256, 0, stream>>>(W2, WH, WL);
    GCN_gemm_bf16x3<<<cdiv(KK1P / 16, 8), B, 0, stream>>>(XN, WH, WL, H, KK1P / 16);
    GCN_fill_f32<<<cdiv(KK1, B), B, 0, stream>>>(DEG, 1.0f, KK1);
    GCN_deg2<<<cdiv(NE, B), B, 0, stream>>>(row, col, R, KK1, DEG, NE);
    GCN_dinv<<<cdiv(KK1, B), B, 0, stream>>>(DEG, KK1);
    GCN_agg_self<<<cdiv((long)KK1 * CC, B), B, 0, stream>>>(H, DEG, A, KK1);
    GCN_agg_edges2<<<cdiv((long)NE * 32, B), B, 0, stream>>>(row, col, R, KK1, H, DEG, A, NE);
    GCN_bias_bn_relu<<<cdiv(K1C, B), B, 0, stream>>>(A, A, b2, g2, be2, m2, v2, KK1, KK1P);

    // ---------------- conv3 (same pooled graph, same dinv) ----------------
    GCN_prep_w<<<4, 256, 0, stream>>>(W3, WH, WL);
    GCN_gemm_bf16x3<<<cdiv(KK1P / 16, 8), B, 0, stream>>>(A, WH, WL, H, KK1P / 16);
    GCN_agg_self<<<cdiv((long)KK1 * CC, B), B, 0, stream>>>(H, DEG, XN, KK1);
    GCN_agg_edges2<<<cdiv((long)NE * 32, B), B, 0, stream>>>(row, col, R, KK1, H, DEG, XN, NE);
    GCN_bias_bn_relu<<<cdiv(K1C, B), B, 0, stream>>>(XN, XN, b3, g3, be3, m3, v3, KK1, KK1P);

    // ---------------- pool2 (top-K2) -> output ----------------
    GCN_scores<<<cdiv((long)KK1 * 32, B), B, 0, stream>>>(XN, p2, PN + 1, S, KEY, KK1);
    GCN_fill_i32<<<cdiv(NBUCKET, B), B, 0, stream>>>(HIST, 0, NBUCKET);
    GCN_fill_i32<<<cdiv(NBUCKET, B), B, 0, stream>>>(CUR, 0, NBUCKET);
    GCN_hist<<<cdiv(KK1, B), B, 0, stream>>>(KEY, HIST, KK1);
    GCN_bucket_sufscan<<<1, 1024, 0, stream>>>(HIST, GT);
    GCN_bucket_scatter<<<cdiv(KK1, B), B, 0, stream>>>(KEY, GT, HIST, CUR, BL, KK1);
    GCN_rank_fast<<<cdiv(KK1, B), B, 0, stream>>>(KEY, GT, HIST, BL, R, KK1);
    GCN_pool_scatter<<<cdiv((long)KK1 * CC, B), B, 0, stream>>>(XN, S, R, KK2, out, KK1);
}
